// Aligner_78563541778597
// MI455X (gfx1250) — matrix operand order repaired and checked by reading, not compiled, not run on hardware
//
#include <hip/hip_runtime.h>
#include <hip/hip_bf16.h>
#include <math.h>

#define B_   64
#define S_   256
#define I_   512
#define H_   80
#define T_   600
#define C_   512
#define C3_  1536
#define LOC_ 8
#define KW_  21
#define P_   11
#define XK_  608   // I+H = 592 padded to multiple of 32
#define N2_  176   // LOC*KW = 168 padded to multiple of 16

typedef __attribute__((ext_vector_type(16))) __bf16          v16bf;
typedef __attribute__((ext_vector_type(16))) unsigned short  v16u;
typedef __attribute__((ext_vector_type(8)))  float           v8f;

__device__ __forceinline__ unsigned short f32_to_bf16(float f) {
  unsigned int u = __float_as_uint(f);
  unsigned int r = 0x7FFFu + ((u >> 16) & 1u);
  return (unsigned short)((u + r) >> 16);
}

__device__ __forceinline__ float sigmoidf_(float x) { return 1.0f / (1.0f + __expf(-x)); }

// ---------------------------------------------------------------------------
// Weight convert + swizzle into WMMA B-fragment order.
// Source W: (Nsrc x Ksrc) f32 row-major; logical B[k][n] = W[n*Ksrc + k].
// Fragment element (kt, nt, lane, j) -> k = kt*32 + (lane>=16?8:0) + (j<8?j:j+8),
//                                       n = nt*16 + (lane&15)
// stored at WB[((kt*NT + nt)*32 + lane)*16 + j], zero-padded out of range.
// ---------------------------------------------------------------------------
template <int NT, int KT, int Nsrc, int Ksrc>
__global__ void convert_swizzle_kernel(const float* __restrict__ W,
                                       unsigned short* __restrict__ WB) {
  int idx = blockIdx.x * 256 + threadIdx.x;
  if (idx >= KT * NT * 512) return;
  int j    = idx & 15;
  int lane = (idx >> 4) & 31;
  int tile = idx >> 9;
  int nt = tile % NT;
  int kt = tile / NT;
  int n = nt * 16 + (lane & 15);
  int k = kt * 32 + ((lane >> 4) ? 8 : 0) + ((j < 8) ? j : (j + 8));
  float v = (n < Nsrc && k < Ksrc) ? W[(size_t)n * Ksrc + k] : 0.0f;
  WB[idx] = f32_to_bf16(v);
}

// ---------------------------------------------------------------------------
// Init: state = 0, alpha = one-hot at s==0
// ---------------------------------------------------------------------------
__global__ void init_kernel(float* __restrict__ state, float* __restrict__ alpha) {
  int idx = blockIdx.x * 256 + threadIdx.x;
  if (idx < B_ * C_) state[idx] = 0.0f;
  if (idx < B_ * S_) alpha[idx] = ((idx & (S_ - 1)) == 0) ? 1.0f : 0.0f;
}

// ---------------------------------------------------------------------------
// build_x: X[b][0:80] = gt[b,t,:]; X[b][80:592] = alpha[b,:] @ enc[b,:,:];
//          X[b][592:608] = 0.   One block (256 threads) per batch.
// ---------------------------------------------------------------------------
__global__ void build_x_kernel(const float* __restrict__ enc,
                               const float* __restrict__ alpha,
                               const float* __restrict__ gt, int t,
                               float* __restrict__ X) {
  int b = blockIdx.x;
  int tid = threadIdx.x;
  __shared__ float al[S_];
  al[tid] = alpha[b * S_ + tid];
  __syncthreads();

  if (tid < H_) X[(size_t)b * XK_ + tid] = gt[((size_t)b * T_ + t) * H_ + tid];
  if (tid < (XK_ - (I_ + H_))) X[(size_t)b * XK_ + (I_ + H_) + tid] = 0.0f;

  for (int i = tid; i < I_; i += 256) {
    const float* e = enc + (size_t)b * S_ * I_ + i;
    float acc = 0.0f;
    #pragma unroll 8
    for (int s = 0; s < S_; ++s) acc += al[s] * e[(size_t)s * I_];
    X[(size_t)b * XK_ + H_ + i] = acc;
  }
}

// ---------------------------------------------------------------------------
// bf16 WMMA GEMM: D(64 x NT*16) = A(64 x K, f32) * B(K x NT*16, swizzled bf16)
// One wave per 16x16 D tile; all strides compile-time. grid=(NT,4), block=32.
// ---------------------------------------------------------------------------
template <int LDA, int NT, int LDD, int K, int ACT, int BIAS>
__global__ void __launch_bounds__(32)
gemm_bf16_wmma(const float* __restrict__ A,
               const unsigned short* __restrict__ Bw,
               float* __restrict__ D,
               const float* __restrict__ bias) {
  const int lane = threadIdx.x;        // 0..31
  const int nt = blockIdx.x;
  const int m0 = blockIdx.y * 16;
  const int hi = lane >> 4;            // 0 or 1
  const int col = nt * 16 + (lane & 15);

  // A: lane (lo) holds K offsets {0..7,16..23}; lane (hi) +8.
  const float* Arow = A + (size_t)(m0 + (lane & 15)) * LDA + (hi ? 8 : 0);
  // B: pre-swizzled fragments, one contiguous v16u per lane per K-chunk.
  const v16u* bp = (const v16u*)Bw + (size_t)nt * 32 + lane;

  constexpr int KT = K / 32;
  v8f acc = {};
  #pragma unroll 2
  for (int kt = 0; kt < KT; ++kt) {
    if (kt + 1 < KT) __builtin_prefetch((const void*)(bp + (size_t)(kt + 1) * NT * 32), 0, 3);
    const float* ap = Arow + kt * 32;
    v16u au;
    #pragma unroll
    for (int j = 0; j < 16; ++j) {
      int kk = (j < 8) ? j : (j + 8);
      au[j] = f32_to_bf16(ap[kk]);
    }
    v16u bu = bp[(size_t)kt * NT * 32];
    v16bf av = __builtin_bit_cast(v16bf, au);
    v16bf bv = __builtin_bit_cast(v16bf, bu);
    acc = __builtin_amdgcn_wmma_f32_16x16x32_bf16(false, av, false, bv,
                                                  (short)0, acc, false, false);
  }

  // D layout: lane L, reg r -> row = r + (L>=16 ? 8 : 0), col = L&15
  int rowbase = m0 + (hi ? 8 : 0);
  float bb = BIAS ? bias[col] : 0.0f;
  #pragma unroll
  for (int r = 0; r < 8; ++r) {
    float v = acc[r] + bb;
    if (ACT) v = tanhf(v);
    D[(size_t)(rowbase + r) * LDD + col] = v;
  }
}

// ---------------------------------------------------------------------------
// GRU gates (biases already folded into gx/gh by the GEMM epilogues).
// ---------------------------------------------------------------------------
__global__ void gru_gates_kernel(const float* __restrict__ gx,
                                 const float* __restrict__ gh,
                                 float* __restrict__ state) {
  int idx = blockIdx.x * 256 + threadIdx.x;   // over B_*C_
  int b = idx >> 9;                           // / C_
  int c = idx & (C_ - 1);
  const float* gxb = gx + (size_t)b * C3_;
  const float* ghb = gh + (size_t)b * C3_;
  float r = sigmoidf_(gxb[c] + ghb[c]);
  float z = sigmoidf_(gxb[C_ + c] + ghb[C_ + c]);
  float n = tanhf(gxb[2 * C_ + c] + r * ghb[2 * C_ + c]);
  float h = state[idx];
  state[idx] = (1.0f - z) * n + z * h;
}

// ---------------------------------------------------------------------------
// Attention: convs + score + prior + masked softmax. One block per batch.
// ---------------------------------------------------------------------------
__global__ void attention_kernel(const float* __restrict__ alpha_in,
                                 const float* __restrict__ dkern,      // B x N2_
                                 const float* __restrict__ loc_conv_w, // 8 x 21
                                 const float* __restrict__ loc_proj_w, // 256 x 8
                                 const float* __restrict__ dyn_w,      // 256 x 8
                                 const float* __restrict__ dyn_b,      // 256
                                 const float* __restrict__ agg_w,      // 256
                                 const float* __restrict__ prior,      // 11
                                 const float* __restrict__ mask,       // B x S
                                 float* __restrict__ alpha_out,        // B x S
                                 float* __restrict__ out, int t) {
  int b = blockIdx.x;
  int s = threadIdx.x;   // 0..255
  __shared__ float al[S_];
  __shared__ float sm[S_];
  al[s] = alpha_in[b * S_ + s];
  __syncthreads();

  float locf[LOC_], dynf[LOC_];
  #pragma unroll
  for (int l = 0; l < LOC_; ++l) { locf[l] = 0.0f; dynf[l] = 0.0f; }
  #pragma unroll
  for (int k = 0; k < KW_; ++k) {
    int sp = s - (KW_ / 2) + k;
    float a = (sp >= 0 && sp < S_) ? al[sp] : 0.0f;
    #pragma unroll
    for (int l = 0; l < LOC_; ++l) {
      locf[l] += loc_conv_w[l * KW_ + k] * a;
      dynf[l] += dkern[(size_t)b * N2_ + l * KW_ + k] * a;
    }
  }

  float energy = 0.0f;
  for (int c = 0; c < C_ / 2; ++c) {
    float sc = dyn_b[c];
    #pragma unroll
    for (int l = 0; l < LOC_; ++l)
      sc += loc_proj_w[c * LOC_ + l] * locf[l] + dyn_w[c * LOC_ + l] * dynf[l];
    energy += agg_w[c] * tanhf(sc);
  }

  float pri = 0.0f;
  #pragma unroll
  for (int k = 0; k < P_; ++k) {
    int sp = s - (P_ - 1) + k;
    if (sp >= 0) pri += prior[k] * al[sp];
  }
  energy += logf(pri + 1e-5f);

  bool valid = mask[b * S_ + s] > 0.0f;

  // masked softmax over S
  sm[s] = valid ? energy : -3.4e38f;
  __syncthreads();
  for (int off = 128; off > 0; off >>= 1) {
    if (s < off) sm[s] = fmaxf(sm[s], sm[s + off]);
    __syncthreads();
  }
  float mx = sm[0];
  __syncthreads();
  float e = valid ? __expf(energy - mx) : 0.0f;
  sm[s] = e;
  __syncthreads();
  for (int off = 128; off > 0; off >>= 1) {
    if (s < off) sm[s] += sm[s + off];
    __syncthreads();
  }
  float a = e / sm[0];

  alpha_out[b * S_ + s] = a;
  out[((size_t)b * T_ + t) * S_ + s] = a;
}

// ---------------------------------------------------------------------------
// Host driver
// ---------------------------------------------------------------------------
extern "C" void kernel_launch(void* const* d_in, const int* in_sizes, int n_in,
                              void* d_out, int out_size, void* d_ws, size_t ws_size,
                              hipStream_t stream) {
  (void)in_sizes; (void)n_in; (void)out_size; (void)ws_size;

  const float* enc        = (const float*)d_in[0];
  const float* mask       = (const float*)d_in[1];
  const float* gt         = (const float*)d_in[2];
  const float* gru_wi     = (const float*)d_in[3];
  const float* gru_wh     = (const float*)d_in[4];
  const float* gru_bi     = (const float*)d_in[5];
  const float* gru_bh     = (const float*)d_in[6];
  const float* loc_conv_w = (const float*)d_in[7];
  const float* loc_proj_w = (const float*)d_in[8];
  const float* kernel_w1  = (const float*)d_in[9];
  const float* kernel_b1  = (const float*)d_in[10];
  const float* kernel_w2  = (const float*)d_in[11];
  const float* dyn_w      = (const float*)d_in[12];
  const float* dyn_b      = (const float*)d_in[13];
  const float* agg_w      = (const float*)d_in[14];
  const float* prior      = (const float*)d_in[15];
  float* out = (float*)d_out;

  // workspace carve (256B aligned)
  char* p = (char*)d_ws;
  auto alloc = [&](size_t bytes) -> void* {
    void* r = (void*)p;
    p += (bytes + 255) & ~(size_t)255;
    return r;
  };
  // tile counts: NT = N/16, KT = Kpad/32
  constexpr int NT_G = C3_ / 16;  // 96
  constexpr int KT_X = XK_ / 32;  // 19
  constexpr int KT_C = C_ / 32;   // 16
  constexpr int NT_1 = C_ / 16;   // 32
  constexpr int NT_2 = N2_ / 16;  // 11

  unsigned short* WiB = (unsigned short*)alloc((size_t)KT_X * NT_G * 512 * 2);
  unsigned short* WhB = (unsigned short*)alloc((size_t)KT_C * NT_G * 512 * 2);
  unsigned short* W1B = (unsigned short*)alloc((size_t)KT_C * NT_1 * 512 * 2);
  unsigned short* W2B = (unsigned short*)alloc((size_t)KT_C * NT_2 * 512 * 2);
  float* X      = (float*)alloc((size_t)B_ * XK_ * 4);
  float* state  = (float*)alloc((size_t)B_ * C_  * 4);
  float* gx     = (float*)alloc((size_t)B_ * C3_ * 4);
  float* gh     = (float*)alloc((size_t)B_ * C3_ * 4);
  float* hidden = (float*)alloc((size_t)B_ * C_  * 4);
  float* dkern  = (float*)alloc((size_t)B_ * N2_ * 4);
  float* alpha  = (float*)alloc((size_t)B_ * S_  * 4);

  // one-time per launch: weight conversion/swizzle + state init (deterministic)
  {
    int tot;
    tot = KT_X * NT_G * 512;
    convert_swizzle_kernel<NT_G, KT_X, C3_, I_ + H_>
        <<<(tot + 255) / 256, 256, 0, stream>>>(gru_wi, WiB);
    tot = KT_C * NT_G * 512;
    convert_swizzle_kernel<NT_G, KT_C, C3_, C_>
        <<<(tot + 255) / 256, 256, 0, stream>>>(gru_wh, WhB);
    tot = KT_C * NT_1 * 512;
    convert_swizzle_kernel<NT_1, KT_C, C_, C_>
        <<<(tot + 255) / 256, 256, 0, stream>>>(kernel_w1, W1B);
    tot = KT_C * NT_2 * 512;
    convert_swizzle_kernel<NT_2, KT_C, LOC_ * KW_, C_>
        <<<(tot + 255) / 256, 256, 0, stream>>>(kernel_w2, W2B);
    init_kernel<<<(B_ * C_ + 255) / 256, 256, 0, stream>>>(state, alpha);
  }

  dim3 gGX(NT_G, B_ / 16);   // (96,4)
  dim3 gM1(NT_1, B_ / 16);   // (32,4)
  dim3 gM2(NT_2, B_ / 16);   // (11,4)

  for (int t = 0; t < T_; ++t) {
    build_x_kernel<<<B_, 256, 0, stream>>>(enc, alpha, gt, t, X);
    gemm_bf16_wmma<XK_, NT_G, C3_, XK_, 0, 1>
        <<<gGX, 32, 0, stream>>>(X, WiB, gx, gru_bi);
    gemm_bf16_wmma<C_, NT_G, C3_, C_, 0, 1>
        <<<gGX, 32, 0, stream>>>(state, WhB, gh, gru_bh);
    gru_gates_kernel<<<(B_ * C_ + 255) / 256, 256, 0, stream>>>(gx, gh, state);
    gemm_bf16_wmma<C_, NT_1, C_, C_, 1, 1>
        <<<gM1, 32, 0, stream>>>(state, W1B, hidden, kernel_b1);
    gemm_bf16_wmma<C_, NT_2, N2_, C_, 0, 0>
        <<<gM2, 32, 0, stream>>>(hidden, W2B, dkern, (const float*)nullptr);
    attention_kernel<<<B_, 256, 0, stream>>>(alpha, dkern, loc_conv_w, loc_proj_w,
                                             dyn_w, dyn_b, agg_w, prior, mask,
                                             alpha, out, t);
  }
}
